// lstm_mlp_50757923504227
// MI455X (gfx1250) — matrix operand order repaired and checked by reading, not compiled, not run on hardware
//
#include <hip/hip_runtime.h>

typedef unsigned short u16;
typedef __attribute__((ext_vector_type(16))) __bf16   v16bf;
typedef __attribute__((ext_vector_type(8)))  float    v8f;
typedef __attribute__((ext_vector_type(4)))  unsigned v4u;

union Frag { v4u u[2]; v16bf v; };

__device__ __forceinline__ u16 f2bf(float f){
  union { float f; unsigned u; } x; x.f = f;
  unsigned u = x.u + 0x7FFFu + ((x.u >> 16) & 1u);
  return (u16)(u >> 16);
}
__device__ __forceinline__ float bf2f(u16 b){
  union { unsigned u; float f; } x; x.u = ((unsigned)b) << 16; return x.f;
}
__device__ __forceinline__ float bits2f(unsigned u){
  union { unsigned u; float f; } x; x.u = u; return x.f;
}
__device__ __forceinline__ unsigned pk2bf(float lo, float hi){
  return (unsigned)f2bf(lo) | ((unsigned)f2bf(hi) << 16);
}
__device__ __forceinline__ float sigf(float x){ return 1.f/(1.f+__expf(-x)); }
__device__ __forceinline__ float tanhfast(float x){
  float e = __expf(-2.f*fabsf(x));
  float r = (1.f-e)/(1.f+e);
  return x < 0.f ? -r : r;
}

// -------- weight prep: f32 [K,N] -> bf16 transposed [N,Kpad], zero-padded, opt. flip ---
__global__ void wt_kernel(const float* __restrict__ W, u16* __restrict__ dst,
                          int K, int N, int Kpad, int flip){
  int idx = blockIdx.x*blockDim.x + threadIdx.x;
  if (idx >= N*Kpad) return;
  int n = idx / Kpad, k = idx % Kpad;
  u16 v = 0;
  if (k < K){
    int ks = flip ? (K-1-k) : k;
    v = f2bf(W[(size_t)ks*N + n]);
  }
  dst[(size_t)n*Kpad + k] = v;
}

// -------- conv1d SAME (k=4, pad 1/2), NWC; out time-major [T*B,32] bf16 (16 padded) ----
__global__ __launch_bounds__(256) void conv_kernel(const float* __restrict__ X,
    const float* __restrict__ cw, const float* __restrict__ cb,
    u16* __restrict__ xbf, int T, int B){
  int m = blockIdx.x*256 + threadIdx.x;      // m = t*B + b
  int t = m / B, b = m % B;
  float acc[16];
#pragma unroll
  for (int co = 0; co < 16; ++co) acc[co] = cb[co];
#pragma unroll
  for (int w = 0; w < 4; ++w){
    int tt = t - 1 + w;
    if (tt >= 0 && tt < T){
#pragma unroll
      for (int ci = 0; ci < 8; ++ci){
        float xv = X[((size_t)b*T + tt)*8 + ci];
#pragma unroll
        for (int co = 0; co < 16; ++co) acc[co] += xv * cw[(w*8+ci)*16 + co];
      }
    }
  }
#pragma unroll
  for (int co = 0; co < 16; ++co) xbf[(size_t)m*32 + co] = f2bf(acc[co]);
#pragma unroll
  for (int co = 16; co < 32; ++co) xbf[(size_t)m*32 + co] = 0;
}

// -------- WMMA GEMM: out = A[M,K]bf16 @ WT[N,K]^T + bias; K % 32 == 0, all unguarded ---
// SW=true: output in WMMA-tile fragment order (one b128 store/lane/tile) for the scan.
// SW=false: row-major bf16 [M,N].
template<int K, int N, bool SW>
__global__ __launch_bounds__(256) void gemm_kernel(const u16* __restrict__ A,
    const u16* __restrict__ WT, const float* __restrict__ bias,
    u16* __restrict__ out, int M){
  (void)M;
  static_assert(K % 32 == 0, "K padded to multiple of 32");
  constexpr int KT = K/32;
  constexpr int NGRP = N/64;                 // 4 n-tiles per wave
  const int wid  = blockIdx.x*8 + (threadIdx.x >> 5);
  const int lane = threadIdx.x & 31;
  const int lr = lane & 15, hi = lane >> 4;
  const int mt = wid / NGRP;
  const int ng = wid % NGRP;
  const int row = mt*16 + lr;

  Frag a[KT];
#pragma unroll
  for (int kt = 0; kt < KT; ++kt){
    const int k0 = kt*32 + hi*16;
    const v4u* p = reinterpret_cast<const v4u*>(A + (size_t)row*K + k0);
    a[kt].u[0] = p[0];
    a[kt].u[1] = p[1];
  }
#pragma unroll
  for (int j = 0; j < 4; ++j){
    const int nt = ng*4 + j;
    const int n = nt*16 + lr;
    v8f acc = {0.f,0.f,0.f,0.f,0.f,0.f,0.f,0.f};
#pragma unroll
    for (int kt = 0; kt < KT; ++kt){
      const int k0 = kt*32 + hi*16;
      const v4u* p = reinterpret_cast<const v4u*>(WT + (size_t)n*K + k0);
      Frag b;
      b.u[0] = p[0];
      b.u[1] = p[1];
      acc = __builtin_amdgcn_wmma_f32_16x16x32_bf16(false, a[kt].v, false, b.v,
                                                    (short)0, acc, false, false);
    }
    const float bv = bias[n];
    if (SW){
      v4u pk;
      pk.x = pk2bf(acc[0]+bv, acc[1]+bv);
      pk.y = pk2bf(acc[2]+bv, acc[3]+bv);
      pk.z = pk2bf(acc[4]+bv, acc[5]+bv);
      pk.w = pk2bf(acc[6]+bv, acc[7]+bv);
      *reinterpret_cast<v4u*>(out + (((size_t)mt*(N/16) + nt)*32 + lane)*8) = pk;
    } else {
#pragma unroll
      for (int v = 0; v < 8; ++v)
        out[(size_t)(mt*16 + v + 8*hi)*N + n] = f2bf(acc[v] + bv);
    }
  }
}

// -------- persistent LSTM scan: 1 block per direction, Wh resident in VGPRs ------------
// xg is in tile-fragment order (see gemm SW=true): C-init is one b128 load per tile.
template<int H>
__global__ __launch_bounds__(512) void scan_kernel(
    const u16* __restrict__ xgA, const u16* __restrict__ xgB,
    const u16* __restrict__ whA, const u16* __restrict__ whB,
    u16* __restrict__ outp, int outStride, int offA, int offB){
  constexpr int T = 4096;
  constexpr int FH = 4*H;
  constexpr int KT = H/32;           // k-tiles of 32
  constexpr int NTW = (FH/16)/16;    // n-tiles per wave (16 waves)
  constexpr int GE = H/16;           // gate elems per thread
  const u16* xg = (blockIdx.x == 0) ? xgA : xgB;
  const u16* wh = (blockIdx.x == 0) ? whA : whB;
  const int off = (blockIdx.x == 0) ? offA : offB;

  __shared__ float zbuf[32][FH];
  __shared__ u16   hbuf[32][H+8];    // +8 halves: rotate LDS banks, keep 16B align

  const int tid = threadIdx.x;
  const int wave = tid >> 5;
  const int lane = tid & 31;
  const int lr = lane & 15, hi = lane >> 4;

  for (int i = tid; i < 32*(H+8); i += 512) ((u16*)hbuf)[i] = 0;

  float creg[GE];
#pragma unroll
  for (int g = 0; g < GE; ++g) creg[g] = 0.f;

  // resident B fragments of Wh (WhT is [4H, H] bf16 row-major)
  Frag bw[NTW][KT];
#pragma unroll
  for (int i = 0; i < NTW; ++i){
    const int n = (wave*NTW + i)*16 + lr;
#pragma unroll
    for (int kt = 0; kt < KT; ++kt){
      const int k0 = kt*32 + hi*16;
      const v4u* p = reinterpret_cast<const v4u*>(wh + (size_t)n*H + k0);
      bw[i][kt].u[0] = p[0];
      bw[i][kt].u[1] = p[1];
    }
  }
  __syncthreads();

  for (int t = 0; t < T; ++t){
    // A fragments from h (bf16) in LDS
    Frag a[2][KT];
#pragma unroll
    for (int mt = 0; mt < 2; ++mt){
      const int r = mt*16 + lr;
#pragma unroll
      for (int kt = 0; kt < KT; ++kt){
        const int k0 = kt*32 + hi*16;
        const v4u* p = reinterpret_cast<const v4u*>(&hbuf[r][k0]);
        a[mt][kt].u[0] = p[0];
        a[mt][kt].u[1] = p[1];
      }
    }
    // z = xg[t] + h @ Wh via WMMA (C initialized from packed xg tiles)
#pragma unroll
    for (int i = 0; i < NTW; ++i){
      const int nt = wave*NTW + i;
      const int ncol = nt*16 + lr;
#pragma unroll
      for (int mt = 0; mt < 2; ++mt){
        const v4u pk = *reinterpret_cast<const v4u*>(
            xg + (((size_t)(2*t + mt)*(FH/16) + nt)*32 + lane)*8);
        v8f acc;
        acc[0] = bits2f(pk.x << 16); acc[1] = bits2f(pk.x & 0xffff0000u);
        acc[2] = bits2f(pk.y << 16); acc[3] = bits2f(pk.y & 0xffff0000u);
        acc[4] = bits2f(pk.z << 16); acc[5] = bits2f(pk.z & 0xffff0000u);
        acc[6] = bits2f(pk.w << 16); acc[7] = bits2f(pk.w & 0xffff0000u);
#pragma unroll
        for (int kt = 0; kt < KT; ++kt)
          acc = __builtin_amdgcn_wmma_f32_16x16x32_bf16(false, a[mt][kt].v, false,
                    bw[i][kt].v, (short)0, acc, false, false);
#pragma unroll
        for (int v = 0; v < 8; ++v)
          zbuf[mt*16 + v + 8*hi][ncol] = acc[v];
      }
    }
    __syncthreads();

    // gates: i,f,g,o at j, j+H, j+2H, j+3H
#pragma unroll
    for (int g = 0; g < GE; ++g){
      const int flat = g*512 + tid;
      const int b = flat / H;
      const int j = flat % H;
      const float zi = zbuf[b][j];
      const float zf = zbuf[b][j+H];
      const float zg = zbuf[b][j+2*H];
      const float zo = zbuf[b][j+3*H];
      float c = sigf(zf)*creg[g] + sigf(zi)*tanhfast(zg);
      float h = sigf(zo)*tanhfast(c);
      creg[g] = c;
      const u16 hb = f2bf(h);
      hbuf[b][j] = hb;
      outp[((size_t)t*32 + b)*outStride + off + j] = hb;
    }
    if (t + 1 < T){
      const char* pf = (const char*)(xg + (size_t)(t+1)*32*FH) + tid*(FH/8);
      __builtin_prefetch(pf, 0, 3);   // global_prefetch next step's xg slice
    }
    __syncthreads();
  }
}

// -------- score finish: s = Va . tanh(pre) + bva  (pre from WMMA GEMM) -----------------
__global__ __launch_bounds__(256) void score_finish_kernel(const u16* __restrict__ pre,
    const float* __restrict__ Va, const float* __restrict__ bva,
    float* __restrict__ score){
  int m = blockIdx.x*256 + threadIdx.x;     // m = t*B + b
  const u16* r = pre + (size_t)m*64;
  float s = bva[0];
#pragma unroll
  for (int j = 0; j < 64; ++j) s += Va[j]*tanhfast(bf2f(r[j]));
  score[m] = s;
}

// -------- softmax over T, context, MLP head; one block per batch ------------------------
__global__ __launch_bounds__(256) void attn_kernel(const u16* __restrict__ enc,
    const float* __restrict__ score,
    const float* __restrict__ Wd1, const float* __restrict__ bd1,
    const float* __restrict__ Wd2, const float* __restrict__ bd2,
    float* __restrict__ out){
  constexpr int T = 4096, B = 32;
  const int b = blockIdx.x, tid = threadIdx.x;
  __shared__ float red[256];
  __shared__ float wbuf[T];
  __shared__ float ctx[256];
  float m = -3.4e38f;
  for (int t = tid; t < T; t += 256) m = fmaxf(m, score[(size_t)t*B + b]);
  red[tid] = m; __syncthreads();
  for (int s = 128; s > 0; s >>= 1){ if (tid < s) red[tid] = fmaxf(red[tid], red[tid+s]); __syncthreads(); }
  m = red[0]; __syncthreads();
  float sum = 0.f;
  for (int t = tid; t < T; t += 256){
    float e = __expf(score[(size_t)t*B + b] - m);
    wbuf[t] = e; sum += e;
  }
  red[tid] = sum; __syncthreads();
  for (int s = 128; s > 0; s >>= 1){ if (tid < s) red[tid] += red[tid+s]; __syncthreads(); }
  const float inv = 1.f/red[0];
  float acc = 0.f;
  for (int t = 0; t < T; ++t) acc += wbuf[t]*bf2f(enc[((size_t)t*B + b)*256 + tid]);
  ctx[tid] = acc*inv;
  __syncthreads();
  if (tid < 64){
    float a = bd1[tid];
    for (int k = 0; k < 256; ++k) a += ctx[k]*Wd1[(size_t)k*64 + tid];
    red[tid] = tanhfast(a);
  }
  __syncthreads();
  if (tid == 0){
    float o = bd2[0];
    for (int k = 0; k < 64; ++k) o += red[k]*Wd2[k];
    out[b] = o;
  }
}

extern "C" void kernel_launch(void* const* d_in, const int* in_sizes, int n_in,
                              void* d_out, int out_size, void* d_ws, size_t ws_size,
                              hipStream_t stream){
  (void)in_sizes; (void)n_in; (void)out_size; (void)ws_size;
  const float* X      = (const float*)d_in[0];
  const float* conv_w = (const float*)d_in[1];
  const float* conv_b = (const float*)d_in[2];
  const float* Wx1 = (const float*)d_in[3];  const float* Wh1 = (const float*)d_in[4];  const float* b1 = (const float*)d_in[5];
  const float* Wx2 = (const float*)d_in[6];  const float* Wh2 = (const float*)d_in[7];  const float* b2 = (const float*)d_in[8];
  const float* Wx3 = (const float*)d_in[9];  const float* Wh3 = (const float*)d_in[10]; const float* b3 = (const float*)d_in[11];
  const float* Wx4 = (const float*)d_in[12]; const float* Wh4 = (const float*)d_in[13]; const float* b4 = (const float*)d_in[14];
  const float* Wa  = (const float*)d_in[15]; const float* ba  = (const float*)d_in[16];
  const float* Va  = (const float*)d_in[17]; const float* bva = (const float*)d_in[18];
  const float* Wd1 = (const float*)d_in[19]; const float* bd1 = (const float*)d_in[20];
  const float* Wd2 = (const float*)d_in[21]; const float* bd2 = (const float*)d_in[22];

  const int B = 32, T = 4096, M = B*T;

  char* w = (char*)d_ws;
  size_t off = 0;
  auto carve = [&](size_t bytes)->char*{
    char* p = w + off; off = (off + bytes + 255) & ~(size_t)255; return p;
  };
  u16* regionA = (u16*)carve((size_t)M*512*2);   // xg1+xg2, later reused as xg3
  u16* xg1 = regionA;
  u16* xg2 = regionA + (size_t)M*256;
  u16* xg3 = regionA;
  u16* xg4 = (u16*)carve((size_t)M*512*2);
  u16*   encb  = (u16*)carve((size_t)M*256*2);   // enc, bf16, time-major [T*B,256]
  u16*   catb  = (u16*)carve((size_t)M*128*2);
  u16*   xbf   = (u16*)carve((size_t)M*32*2);    // conv out, K padded 16->32
  u16*   spre  = (u16*)carve((size_t)M*64*2);    // attention pre-activation [T*B,64]
  float* score = (float*)carve((size_t)M*4);
  u16* WxT1 = (u16*)carve((size_t)256*32*2);
  u16* WxT2 = (u16*)carve((size_t)256*32*2);
  u16* WxT3 = (u16*)carve((size_t)512*128*2);
  u16* WxT4 = (u16*)carve((size_t)512*128*2);
  u16* WhT1 = (u16*)carve((size_t)256*64*2);
  u16* WhT2 = (u16*)carve((size_t)256*64*2);
  u16* WhT3 = (u16*)carve((size_t)512*128*2);
  u16* WhT4 = (u16*)carve((size_t)512*128*2);
  u16* WaT  = (u16*)carve((size_t)64*256*2);

  // weight prep (flip folds feature-axis flip of LSTM2/4 inputs into Wx)
  wt_kernel<<<(256*32+255)/256, 256, 0, stream>>>(Wx1, WxT1, 16, 256, 32, 0);
  wt_kernel<<<(256*32+255)/256, 256, 0, stream>>>(Wx2, WxT2, 16, 256, 32, 1);
  wt_kernel<<<(512*128+255)/256, 256, 0, stream>>>(Wx3, WxT3, 128, 512, 128, 0);
  wt_kernel<<<(512*128+255)/256, 256, 0, stream>>>(Wx4, WxT4, 128, 512, 128, 1);
  wt_kernel<<<(256*64+255)/256, 256, 0, stream>>>(Wh1, WhT1, 64, 256, 64, 0);
  wt_kernel<<<(256*64+255)/256, 256, 0, stream>>>(Wh2, WhT2, 64, 256, 64, 0);
  wt_kernel<<<(512*128+255)/256, 256, 0, stream>>>(Wh3, WhT3, 128, 512, 128, 0);
  wt_kernel<<<(512*128+255)/256, 256, 0, stream>>>(Wh4, WhT4, 128, 512, 128, 0);
  wt_kernel<<<(64*256+255)/256, 256, 0, stream>>>(Wa, WaT, 256, 64, 256, 0);

  conv_kernel<<<M/256, 256, 0, stream>>>(X, conv_w, conv_b, xbf, T, B);

  gemm_kernel<32,256,true><<<(M/16)*(256/64)/8, 256, 0, stream>>>(xbf, WxT1, b1, xg1, M);
  gemm_kernel<32,256,true><<<(M/16)*(256/64)/8, 256, 0, stream>>>(xbf, WxT2, b2, xg2, M);

  scan_kernel<64><<<2, 512, 0, stream>>>(xg1, xg2, WhT1, WhT2, catb, 128, 0, 64);

  gemm_kernel<128,512,true><<<(M/16)*(512/64)/8, 256, 0, stream>>>(catb, WxT3, b3, xg3, M);
  gemm_kernel<128,512,true><<<(M/16)*(512/64)/8, 256, 0, stream>>>(catb, WxT4, b4, xg4, M);

  scan_kernel<128><<<2, 512, 0, stream>>>(xg3, xg4, WhT3, WhT4, encb, 256, 0, 128);

  // attention: pre = enc @ Wa + ba via WMMA, then tanh/Va reduction, softmax+ctx+MLP
  gemm_kernel<256,64,false><<<(M/16)*(64/64)/8, 256, 0, stream>>>(encb, WaT, ba, spre, M);
  score_finish_kernel<<<M/256, 256, 0, stream>>>(spre, Va, bva, score);
  attn_kernel<<<32, 256, 0, stream>>>(encb, score, Wd1, bd1, Wd2, bd2, (float*)d_out);
}
